// Model_91328184582627
// MI455X (gfx1250) — matrix operand order repaired and checked by reading, not compiled, not run on hardware
//
#include <hip/hip_runtime.h>

// ---------------------------------------------------------------------------
// Seq2seq GRU encoder/decoder with Bahdanau attention, CDNA5 (gfx1250).
// GEMMs: V_WMMA_F32_16X16X32_BF16 (bf16 in, fp32 accumulate).
// Weight tiles staged to LDS via GLOBAL_LOAD_ASYNC_TO_LDS (ASYNCcnt path),
// double-buffered so the async copy of chunk c+1 overlaps the WMMA compute
// of chunk c. One workgroup barrier per K-chunk.
// ---------------------------------------------------------------------------

#if defined(__has_builtin)
#  if __has_builtin(__builtin_amdgcn_global_load_async_to_lds_b128)
#    define ASYNC_LDS 1
#  endif
#endif
#ifndef ASYNC_LDS
#  define ASYNC_LDS 0
#endif

namespace {

constexpr int kS = 512;   // source length
constexpr int kT = 256;   // target length
constexpr int kB = 64;    // batch
constexpr int kE = 512;   // encoder input dim
constexpr int kD = 512;   // decoder input dim
constexpr int kH = 1024;  // hidden
constexpr int kO = 512;   // output dim

constexpr int KCH  = 128;      // K-chunk staged to LDS (4 WMMA k-steps)
constexpr int BPAD = KCH + 8;  // padded LDS row stride (bf16) -> 272B rows

typedef __bf16 bf16_t;
typedef __attribute__((ext_vector_type(16))) __bf16 v16bf;
typedef __attribute__((ext_vector_type(8)))  __bf16 v8bf;
typedef __attribute__((ext_vector_type(8)))  float  v8f;
typedef int v4i_ __attribute__((vector_size(16)));  // b128 payload type

#if ASYNC_LDS
typedef __attribute__((address_space(1))) v4i_* gptr128_t;  // global b128
typedef __attribute__((address_space(3))) v4i_* lptr128_t;  // LDS b128
#endif

__device__ __forceinline__ void wait_async0() {
#if defined(__has_builtin) && __has_builtin(__builtin_amdgcn_s_wait_asynccnt)
  __builtin_amdgcn_s_wait_asynccnt(0);
#else
  asm volatile("s_wait_asynccnt 0x0" ::: "memory");
#endif
}

// --- WMMA tile loaders (ISA 7.12.2 16-bit layouts, wave32) -----------------
// A (16x32, MxK): lanes 0-15 = rows M0..15 holding K 0..7 / 16..23;
// lanes 16-31 = same rows holding K 8..15 / 24..31.
__device__ __forceinline__ v16bf load_A_tile(const bf16_t* __restrict__ X,
                                             int m0, int k0, int ldk, int lane) {
  const int m  = m0 + (lane & 15);
  const int kb = k0 + ((lane >> 4) << 3);
  const bf16_t* p = X + (size_t)m * ldk + kb;
  v8bf lo = *reinterpret_cast<const v8bf*>(p);
  v8bf hi = *reinterpret_cast<const v8bf*>(p + 16);
  return __builtin_shufflevector(lo, hi, 0, 1, 2, 3, 4, 5, 6, 7,
                                 8, 9, 10, 11, 12, 13, 14, 15);
}

// B fragment from LDS-staged tile (16 rows x KCH cols, row stride BPAD).
// lanes 0-15 = cols N0..15 holding K kk+0..7 / kk+16..23; lanes 16-31 hold kk+8..15 / kk+24..31.
// Row stride 272B => bank group 4*row, no 16-way conflicts; 16B aligned.
__device__ __forceinline__ v16bf load_B_lds(const bf16_t* base, int kk, int lane) {
  const bf16_t* p = base + (size_t)(lane & 15) * BPAD + kk + ((lane >> 4) << 3);
  v8bf lo = *reinterpret_cast<const v8bf*>(p);
  v8bf hi = *reinterpret_cast<const v8bf*>(p + 16);
  return __builtin_shufflevector(lo, hi, 0, 1, 2, 3, 4, 5, 6, 7,
                                 8, 9, 10, 11, 12, 13, 14, 15);
}

// Cooperative copy of a 16 x KCH bf16 weight tile into LDS (async, no wait).
// 16 rows * (KCH/8) 16-byte chunks = 256 chunks; 128 threads -> 2 each.
__device__ __forceinline__ void stage_tile(const bf16_t* __restrict__ src,
                                           int ldk, bf16_t* __restrict__ dst,
                                           int tid) {
  constexpr int CH = 16 * (KCH / 8);
  for (int c = tid; c < CH; c += 128) {
    const int r  = c >> 4;          // chunk row
    const int k8 = (c & 15) << 3;   // bf16 offset within row
#if ASYNC_LDS
    __builtin_amdgcn_global_load_async_to_lds_b128(
        (gptr128_t)(src + (size_t)r * ldk + k8),
        (lptr128_t)(dst + r * BPAD + k8),
        0, 0);
#else
    *reinterpret_cast<v8bf*>(dst + r * BPAD + k8) =
        *reinterpret_cast<const v8bf*>(src + (size_t)r * ldk + k8);
#endif
  }
}

__device__ __forceinline__ v8f wmma_bf16(v16bf a, v16bf b, v8f c) {
  return __builtin_amdgcn_wmma_f32_16x16x32_bf16(
      /*neg_a=*/false, a, /*neg_b=*/false, b,
      /*c_mod=*/(short)0, c, /*reuse_a=*/false, /*reuse_b=*/false);
}

__device__ __forceinline__ float sigmoid_f(float x) {
  return 1.0f / (1.0f + __expf(-x));
}

// --- utility kernels --------------------------------------------------------
__global__ void cvt_f32_to_bf16(const float* __restrict__ src,
                                bf16_t* __restrict__ dst, size_t n) {
  size_t i = (size_t)blockIdx.x * blockDim.x + threadIdx.x;
  if (i < n) dst[i] = (bf16_t)src[i];
}

__global__ void fill_zero_f32(float* __restrict__ p, int n) {
  int i = blockIdx.x * blockDim.x + threadIdx.x;
  if (i < n) p[i] = 0.0f;
}

__global__ void fill_zero_bf16(bf16_t* __restrict__ p, int n) {
  int i = blockIdx.x * blockDim.x + threadIdx.x;
  if (i < n) p[i] = (bf16_t)0.0f;
}

// --- fused GRU step (software-pipelined) ------------------------------------
// grid = H/16 blocks (column slice), block = 128 threads = 4 waves (wave w
// owns batch rows w*16..w*16+15). The two GEMMs (X@Wih^T over Kx, then
// Hin@Whh^T over H) are fused into one chunk sequence; weight tiles for the
// 3 gates are async-staged into a double-buffered LDS tile so the copy of
// chunk c+1 overlaps the 12 WMMAs of chunk c. One barrier per chunk.
__global__ void __launch_bounds__(128)
gru_step_kernel(const bf16_t* __restrict__ Xbf, int Kx,
                const bf16_t* __restrict__ Hin_bf,
                const bf16_t* __restrict__ Wih,
                const bf16_t* __restrict__ Whh,
                const float* __restrict__ bih, const float* __restrict__ bhh,
                float* __restrict__ Hf, bf16_t* __restrict__ Hout_bf,
                float* __restrict__ enc_out_t) {
  __shared__ bf16_t sB[2][3][16][BPAD];  // ~25.5 KB, double-buffered

  const int tid  = threadIdx.x;
  const int lane = tid & 31;
  const int wave = tid >> 5;
  const int n0 = blockIdx.x * 16;  // column slice within H
  const int m0 = wave * 16;        // batch-row tile

  const int nch1 = Kx / KCH;           // chunks of the Wih GEMM
  const int nch  = nch1 + kH / KCH;    // + chunks of the Whh GEMM

  auto stage_chunk = [&](int c, int buf) {
    const bf16_t* W;
    int ld, kc;
    if (c < nch1) { W = Wih; ld = Kx; kc = c * KCH; }
    else          { W = Whh; ld = kH; kc = (c - nch1) * KCH; }
    stage_tile(W + (size_t)(n0)*ld + kc,            ld, &sB[buf][0][0][0], tid);
    stage_tile(W + (size_t)(kH + n0) * ld + kc,     ld, &sB[buf][1][0][0], tid);
    stage_tile(W + (size_t)(2 * kH + n0) * ld + kc, ld, &sB[buf][2][0][0], tid);
  };

  v8f acc_r = {}, acc_z = {}, acc_in = {}, acc_hn = {};

  stage_chunk(0, 0);
  for (int c = 0; c < nch; ++c) {
    const int buf = c & 1;
#if ASYNC_LDS
    wait_async0();       // chunk c landed in LDS (only chunk in flight)
#endif
    __syncthreads();     // visible to all waves; all done computing chunk c-1
    if (c + 1 < nch) stage_chunk(c + 1, buf ^ 1);  // overlap with compute

    if (c < nch1) {      // X @ Wih^T : r, z, i_n
      const int kc = c * KCH;
#pragma unroll
      for (int kk = 0; kk < KCH; kk += 32) {
        v16bf a  = load_A_tile(Xbf, m0, kc + kk, Kx, lane);
        v16bf br = load_B_lds(&sB[buf][0][0][0], kk, lane);
        v16bf bz = load_B_lds(&sB[buf][1][0][0], kk, lane);
        v16bf bn = load_B_lds(&sB[buf][2][0][0], kk, lane);
        acc_r  = wmma_bf16(a, br, acc_r);
        acc_z  = wmma_bf16(a, bz, acc_z);
        acc_in = wmma_bf16(a, bn, acc_in);
      }
    } else {             // Hin @ Whh^T : r, z, h_n
      const int kc = (c - nch1) * KCH;
#pragma unroll
      for (int kk = 0; kk < KCH; kk += 32) {
        v16bf a  = load_A_tile(Hin_bf, m0, kc + kk, kH, lane);
        v16bf br = load_B_lds(&sB[buf][0][0][0], kk, lane);
        v16bf bz = load_B_lds(&sB[buf][1][0][0], kk, lane);
        v16bf bn = load_B_lds(&sB[buf][2][0][0], kk, lane);
        acc_r  = wmma_bf16(a, br, acc_r);
        acc_z  = wmma_bf16(a, bz, acc_z);
        acc_hn = wmma_bf16(a, bn, acc_hn);
      }
    }
  }

  // epilogue: C/D layout -> (M,N): reg v, lane<16 => M=m0+v, lane>=16 => M=m0+8+v
  const int ncol = n0 + (lane & 15);
  const int mrow = m0 + ((lane >> 4) << 3);
  const float b_r  = bih[ncol] + bhh[ncol];
  const float b_z  = bih[kH + ncol] + bhh[kH + ncol];
  const float b_in = bih[2 * kH + ncol];
  const float b_hn = bhh[2 * kH + ncol];
#pragma unroll
  for (int v = 0; v < 8; ++v) {
    const size_t idx = (size_t)(mrow + v) * kH + ncol;
    const float r = sigmoid_f(acc_r[v] + b_r);
    const float z = sigmoid_f(acc_z[v] + b_z);
    const float n = tanhf(acc_in[v] + b_in + r * (acc_hn[v] + b_hn));
    const float hold = Hf[idx];
    const float hnew = (1.0f - z) * n + z * hold;
    Hf[idx] = hnew;
    Hout_bf[idx] = (bf16_t)hnew;
    if (enc_out_t) enc_out_t[idx] = hnew;
  }
}

// --- Y = [X1 | X2] @ W^T + bias (optional ReLU), WMMA, pipelined -----------
// grid = N/16, block = 128 (4 waves over the 64 batch rows). K1 is a
// multiple of KCH, so the concatenated weight columns are chunk-aligned
// and a single chunk index covers both halves.
__global__ void __launch_bounds__(128)
linear_cat_kernel(const bf16_t* __restrict__ X1, int K1,
                  const bf16_t* __restrict__ X2, int K2,
                  const bf16_t* __restrict__ W,
                  const float* __restrict__ bias,
                  float* __restrict__ Yf, bf16_t* __restrict__ Ybf,
                  int N, int relu) {
  __shared__ bf16_t sB[2][16][BPAD];  // ~8.5 KB, double-buffered

  const int tid  = threadIdx.x;
  const int lane = tid & 31;
  const int wave = tid >> 5;
  const int n0 = blockIdx.x * 16;
  const int m0 = wave * 16;
  const int ldw = K1 + K2;

  const int nch1 = K1 / KCH;
  const int nch  = nch1 + K2 / KCH;

  v8f acc = {};
  stage_tile(W + (size_t)n0 * ldw, ldw, &sB[0][0][0], tid);  // chunk 0
  for (int c = 0; c < nch; ++c) {
    const int buf = c & 1;
#if ASYNC_LDS
    wait_async0();
#endif
    __syncthreads();
    if (c + 1 < nch)
      stage_tile(W + (size_t)n0 * ldw + (c + 1) * KCH, ldw, &sB[buf ^ 1][0][0], tid);

    const bf16_t* Asrc = (c < nch1) ? X1 : X2;
    const int lda = (c < nch1) ? K1 : K2;
    const int kc  = ((c < nch1) ? c : c - nch1) * KCH;
#pragma unroll
    for (int kk = 0; kk < KCH; kk += 32) {
      v16bf a = load_A_tile(Asrc, m0, kc + kk, lda, lane);
      v16bf b = load_B_lds(&sB[buf][0][0], kk, lane);
      acc = wmma_bf16(a, b, acc);
    }
  }

  const int ncol = n0 + (lane & 15);
  const int mrow = m0 + ((lane >> 4) << 3);
  const float bv = bias ? bias[ncol] : 0.0f;
#pragma unroll
  for (int v = 0; v < 8; ++v) {
    float y = acc[v] + bv;
    if (relu) y = fmaxf(y, 0.0f);
    const size_t idx = (size_t)(mrow + v) * N + ncol;
    if (Yf)  Yf[idx] = y;
    if (Ybf) Ybf[idx] = (bf16_t)y;
  }
}

// --- row softmax (B rows of length `cols`) ---------------------------------
__global__ void __launch_bounds__(256)
softmax_rows(float* __restrict__ A, int cols) {
  __shared__ float red[256];
  const int tid = threadIdx.x;
  float* row = A + (size_t)blockIdx.x * cols;

  float m = -3.4e38f;
  for (int c = tid; c < cols; c += 256) m = fmaxf(m, row[c]);
  red[tid] = m;
  __syncthreads();
  for (int s = 128; s > 0; s >>= 1) {
    if (tid < s) red[tid] = fmaxf(red[tid], red[tid + s]);
    __syncthreads();
  }
  m = red[0];
  __syncthreads();

  float sum = 0.0f;
  for (int c = tid; c < cols; c += 256) {
    float e = __expf(row[c] - m);
    row[c] = e;
    sum += e;
  }
  red[tid] = sum;
  __syncthreads();
  for (int s = 128; s > 0; s >>= 1) {
    if (tid < s) red[tid] += red[tid + s];
    __syncthreads();
  }
  const float inv = 1.0f / red[0];
  for (int c = tid; c < cols; c += 256) row[c] *= inv;
}

// --- ctx[b,h] = sum_s w[b,s] * enc_out[s,b,h] ------------------------------
// Per-batch matvec (B matrix differs per b => no WMMA). enc_out (134 MB)
// is L2-resident on MI455X (192 MB L2), so this streams from L2.
__global__ void __launch_bounds__(1024)
ctx_kernel(const float* __restrict__ w, const float* __restrict__ enc,
           float* __restrict__ ctxf, bf16_t* __restrict__ ctxbf) {
  const int b = blockIdx.x;
  const int h = threadIdx.x;
  const float* wrow = w + (size_t)b * kS;
  float acc = 0.0f;
  for (int s = 0; s < kS; ++s) {
    if ((s & 7) == 0 && s + 16 < kS)
      __builtin_prefetch(enc + ((size_t)(s + 16) * kB + b) * kH + h, 0, 1);
    acc = fmaf(wrow[s], enc[((size_t)s * kB + b) * kH + h], acc);
  }
  ctxf[(size_t)b * kH + h] = acc;
  ctxbf[(size_t)b * kH + h] = (bf16_t)acc;
}

}  // namespace

// ---------------------------------------------------------------------------
extern "C" void kernel_launch(void* const* d_in, const int* in_sizes, int n_in,
                              void* d_out, int out_size, void* d_ws, size_t ws_size,
                              hipStream_t stream) {
  (void)in_sizes; (void)n_in; (void)out_size; (void)ws_size;

  const float* input_tensor = (const float*)d_in[0];   // [S,B,E]
  const float* target_tensor = (const float*)d_in[1];  // [T,B,D]
  const float* enc_Wih = (const float*)d_in[2];        // [3H,E]
  const float* enc_Whh = (const float*)d_in[3];        // [3H,H]
  const float* enc_bih = (const float*)d_in[4];
  const float* enc_bhh = (const float*)d_in[5];
  const float* attn_W = (const float*)d_in[6];         // [S, D+H]
  const float* attn_b = (const float*)d_in[7];
  const float* comb_W = (const float*)d_in[8];         // [H, D+H]
  const float* comb_b = (const float*)d_in[9];
  const float* dec_Wih = (const float*)d_in[10];       // [3H,H]
  const float* dec_Whh = (const float*)d_in[11];       // [3H,H]
  const float* dec_bih = (const float*)d_in[12];
  const float* dec_bhh = (const float*)d_in[13];
  const float* out_W = (const float*)d_in[14];         // [O,H]
  const float* out_b = (const float*)d_in[15];
  float* out = (float*)d_out;                          // [T,B,O]

  // --- workspace bump allocator (256B aligned) ---
  char* base = (char*)d_ws;
  size_t off = 0;
  auto bump = [&](size_t bytes) -> void* {
    void* p = base + off;
    off += (bytes + 255) & ~(size_t)255;
    return p;
  };

  bf16_t* encWih_bf = (bf16_t*)bump((size_t)3 * kH * kE * 2);
  bf16_t* encWhh_bf = (bf16_t*)bump((size_t)3 * kH * kH * 2);
  bf16_t* attnW_bf  = (bf16_t*)bump((size_t)kS * (kD + kH) * 2);
  bf16_t* combW_bf  = (bf16_t*)bump((size_t)kH * (kD + kH) * 2);
  bf16_t* decWih_bf = (bf16_t*)bump((size_t)3 * kH * kH * 2);
  bf16_t* decWhh_bf = (bf16_t*)bump((size_t)3 * kH * kH * 2);
  bf16_t* outW_bf   = (bf16_t*)bump((size_t)kO * kH * 2);
  bf16_t* inp_bf    = (bf16_t*)bump((size_t)kS * kB * kE * 2);
  bf16_t* tgt_bf    = (bf16_t*)bump((size_t)kT * kB * kD * 2);
  float*  enc_out   = (float*)bump((size_t)kS * kB * kH * 4);  // 134 MB, L2-resident
  float*  hF        = (float*)bump((size_t)kB * kH * 4);
  bf16_t* hbfA      = (bf16_t*)bump((size_t)kB * kH * 2);
  bf16_t* hbfB      = (bf16_t*)bump((size_t)kB * kH * 2);
  float*  attn_log  = (float*)bump((size_t)kB * kS * 4);
  float*  ctxF      = (float*)bump((size_t)kB * kH * 4);
  bf16_t* ctx_bf    = (bf16_t*)bump((size_t)kB * kH * 2);
  bf16_t* g_bf      = (bf16_t*)bump((size_t)kB * kH * 2);

  auto cvt = [&](const float* src, bf16_t* dst, size_t n) {
    int grid = (int)((n + 255) / 256);
    cvt_f32_to_bf16<<<grid, 256, 0, stream>>>(src, dst, n);
  };

  // one-time conversions (weights stay L2-resident in bf16, ~28 MB total)
  cvt(enc_Wih, encWih_bf, (size_t)3 * kH * kE);
  cvt(enc_Whh, encWhh_bf, (size_t)3 * kH * kH);
  cvt(attn_W, attnW_bf, (size_t)kS * (kD + kH));
  cvt(comb_W, combW_bf, (size_t)kH * (kD + kH));
  cvt(dec_Wih, decWih_bf, (size_t)3 * kH * kH);
  cvt(dec_Whh, decWhh_bf, (size_t)3 * kH * kH);
  cvt(out_W, outW_bf, (size_t)kO * kH);
  cvt(input_tensor, inp_bf, (size_t)kS * kB * kE);
  cvt(target_tensor, tgt_bf, (size_t)kT * kB * kD);

  // h0 = 0
  fill_zero_f32<<<(kB * kH + 255) / 256, 256, 0, stream>>>(hF, kB * kH);
  fill_zero_bf16<<<(kB * kH + 255) / 256, 256, 0, stream>>>(hbfA, kB * kH);

  bf16_t* hcur = hbfA;  // bf16 mirror of current h (double-buffered)
  bf16_t* hnxt = hbfB;

  // ---- encoder: 512 sequential GRU steps ----
  for (int t = 0; t < kS; ++t) {
    gru_step_kernel<<<kH / 16, 128, 0, stream>>>(
        inp_bf + (size_t)t * kB * kE, kE, hcur, encWih_bf, encWhh_bf,
        enc_bih, enc_bhh, hF, hnxt, enc_out + (size_t)t * kB * kH);
    bf16_t* tmp = hcur; hcur = hnxt; hnxt = tmp;
  }

  // ---- decoder: 256 steps of attention + GRU + output ----
  for (int t = 0; t < kT; ++t) {
    const bf16_t* xt = tgt_bf + (size_t)t * kB * kD;

    // attn_logits = [x_t, h] @ attn_W^T + attn_b      [B, S]
    linear_cat_kernel<<<kS / 16, 128, 0, stream>>>(
        xt, kD, hcur, kH, attnW_bf, attn_b, attn_log, nullptr, kS, 0);
    softmax_rows<<<kB, 256, 0, stream>>>(attn_log, kS);

    // ctx = einsum('bs,sbh->bh', attn_w, enc_out)
    ctx_kernel<<<kB, kH, 0, stream>>>(attn_log, enc_out, ctxF, ctx_bf);

    // g = relu([x_t, ctx] @ comb_W^T + comb_b)        [B, H]
    linear_cat_kernel<<<kH / 16, 128, 0, stream>>>(
        xt, kD, ctx_bf, kH, combW_bf, comb_b, nullptr, g_bf, kH, 1);

    // h = GRU(g, h)
    gru_step_kernel<<<kH / 16, 128, 0, stream>>>(
        g_bf, kH, hcur, decWih_bf, decWhh_bf, dec_bih, dec_bhh,
        hF, hnxt, nullptr);

    // out_t = h @ out_W^T + out_b                     [B, O]
    linear_cat_kernel<<<kO / 16, 128, 0, stream>>>(
        hnxt, kH, nullptr, 0, outW_bf, out_b,
        out + (size_t)t * kB * kO, nullptr, kO, 0);

    bf16_t* tmp = hcur; hcur = hnxt; hnxt = tmp;
  }
}
